// SLAM_Layer_53472342835621
// MI455X (gfx1250) — matrix operand order repaired and checked by reading, not compiled, not run on hardware
//
#include <hip/hip_runtime.h>
#include <hip/hip_bf16.h>

// SLAM layer: h = relu(x_aux@W1+b1); S = h@W2+b2; out[t] = x_main[-1,t,0] @ S[:,t,:]
// N=4, T=1024, M=128, A=64, D=128; f32 in/out, bf16 WMMA internally.

typedef __attribute__((ext_vector_type(16))) __bf16 v16bf;
typedef __attribute__((ext_vector_type(8)))  float  v8f;

// Raw vector types for gfx1250 data-mover builtins
typedef int          v4i_raw __attribute__((vector_size(16)));
typedef unsigned int v4u_raw __attribute__((vector_size(16)));
typedef int          v8i_raw __attribute__((vector_size(32)));

#define TT 1024
#define MM 128
#define AA 64
#define DD 128

#if defined(__gfx1250__) && __has_builtin(__builtin_amdgcn_tensor_load_to_lds)
#define HAVE_TDM 1
#endif
#if defined(__gfx1250__) && __has_builtin(__builtin_amdgcn_global_load_async_to_lds_b128)
#define HAVE_ASYNC_LDS 1
typedef __attribute__((address_space(1))) v4i_raw* gv4i_p;  // global
typedef __attribute__((address_space(3))) v4i_raw* lv4i_p;  // LDS
#endif

__device__ __forceinline__ v8f wmma_bf16(v16bf a, v16bf b, v8f c) {
  return __builtin_amdgcn_wmma_f32_16x16x32_bf16(false, a, false, b, (short)0, c,
                                                 false, false);
}

// Build a v16bf fragment from two 16-byte chunks (elements 0-7 at plo, 8-15 at phi).
__device__ __forceinline__ v16bf frag_from(const __bf16* plo, const __bf16* phi) {
  union { uint4 q[2]; v16bf v; } u;
  u.q[0] = *(const uint4*)plo;
  u.q[1] = *(const uint4*)phi;
  return u.v;
}

// Same, but converting from f32 memory (two 8-float runs).
__device__ __forceinline__ v16bf frag_from_f32(const float* plo, const float* phi) {
  float4 a0 = *(const float4*)plo, a1 = *(const float4*)(plo + 4);
  float4 b0 = *(const float4*)phi, b1 = *(const float4*)(phi + 4);
  v16bf v;
  v[0] = (__bf16)a0.x; v[1] = (__bf16)a0.y; v[2]  = (__bf16)a0.z; v[3]  = (__bf16)a0.w;
  v[4] = (__bf16)a1.x; v[5] = (__bf16)a1.y; v[6]  = (__bf16)a1.z; v[7]  = (__bf16)a1.w;
  v[8] = (__bf16)b0.x; v[9] = (__bf16)b0.y; v[10] = (__bf16)b0.z; v[11] = (__bf16)b0.w;
  v[12] = (__bf16)b1.x; v[13] = (__bf16)b1.y; v[14] = (__bf16)b1.z; v[15] = (__bf16)b1.w;
  return v;
}

// ---------------------------------------------------------------------------
// Kernel 0: one-time transpose+convert of W1 (64x128) and W2 (128x128) into
// bf16 N-major layout so B fragments become contiguous 16B loads.
// ---------------------------------------------------------------------------
__global__ __launch_bounds__(128) void
prep_weights(const float* __restrict__ W1, const float* __restrict__ W2,
             __bf16* __restrict__ W1t, __bf16* __restrict__ W2t) {
  const int n = threadIdx.x;  // output row (N index), 128 threads
  for (int a = 0; a < AA; ++a) W1t[n * AA + a] = (__bf16)W1[a * DD + n];
  for (int d = 0; d < DD; ++d) W2t[n * DD + d] = (__bf16)W2[d * MM + n];
}

// ---------------------------------------------------------------------------
// Kernel 1: fused MLP over rows (m, t); S written to (t, j=m, k) bf16.
// 1 wave = fixed m, 16 consecutive t; 4 waves / block.
// ---------------------------------------------------------------------------
__global__ __launch_bounds__(128) void
slam_mlp_stage(const float* __restrict__ x_aux, const float* __restrict__ b1,
               const float* __restrict__ b2, const __bf16* __restrict__ W1t,
               const __bf16* __restrict__ W2t, __bf16* __restrict__ S_perm)
{
  __shared__ __bf16 h_lds[4][16 * DD];   // 16 KB: per-wave 16x128 h tile
  const int lane = threadIdx.x & 31;
  const int wave = threadIdx.x >> 5;
  const int hi   = lane >> 4;
  const int lrow = lane & 15;
  const int m    = blockIdx.y * 4 + wave;
  const int t0   = blockIdx.x * 16;

  // ---- stage 1: h[16x128] = relu(Xa[16x64] @ W1 + b1) ----
  v8f acc[8];
#pragma unroll
  for (int n = 0; n < 8; ++n) {
    float bias = b1[n * 16 + lrow];
#pragma unroll
    for (int i = 0; i < 8; ++i) acc[n][i] = bias;
  }

  const float* xa = x_aux + ((size_t)m * TT + t0 + lrow) * AA;
#pragma unroll
  for (int kb = 0; kb < 2; ++kb) {
    v16bf af = frag_from_f32(xa + kb * 32 + hi * 8, xa + kb * 32 + 16 + hi * 8);
#pragma unroll
    for (int n = 0; n < 8; ++n) {
      const __bf16* wp = W1t + (n * 16 + lrow) * AA + kb * 32 + hi * 8;
      acc[n] = wmma_bf16(af, frag_from(wp, wp + 16), acc[n]);
    }
  }

  // relu, stash h (bf16, row-major 16x128) in this wave's LDS slab
#pragma unroll
  for (int n = 0; n < 8; ++n) {
#pragma unroll
    for (int i = 0; i < 8; ++i) {
      float v = acc[n][i];
      v = v > 0.f ? v : 0.f;
      h_lds[wave][(i + (hi << 3)) * DD + n * 16 + lrow] = (__bf16)v;
    }
  }

  // ---- stage 2: S[16x128] = h @ W2 + b2 ----
  v8f acc2[8];
#pragma unroll
  for (int n = 0; n < 8; ++n) {
    float bias = b2[n * 16 + lrow];
#pragma unroll
    for (int i = 0; i < 8; ++i) acc2[n][i] = bias;
  }

  v16bf afr[4];
#pragma unroll
  for (int kb = 0; kb < 4; ++kb) {
    const __bf16* hp = &h_lds[wave][lrow * DD + kb * 32 + hi * 8];
    afr[kb] = frag_from(hp, hp + 16);
  }
#pragma unroll
  for (int n = 0; n < 8; ++n) {
#pragma unroll
    for (int kb = 0; kb < 4; ++kb) {
      const __bf16* wp = W2t + (n * 16 + lrow) * DD + kb * 32 + hi * 8;
      acc2[n] = wmma_bf16(afr[kb], frag_from(wp, wp + 16), acc2[n]);
    }
  }

  // scatter S rows into (t, j=m, k) bf16 layout (16 lanes -> 32B runs per t)
#pragma unroll
  for (int n = 0; n < 8; ++n) {
#pragma unroll
    for (int i = 0; i < 8; ++i) {
      int t = t0 + i + (hi << 3);
      S_perm[(size_t)t * (MM * MM) + m * MM + n * 16 + lrow] = (__bf16)acc2[n][i];
    }
  }
}

// ---------------------------------------------------------------------------
// Kernel 2: out[t] = x_main_last[t] (128x128 f32) @ S_perm[t] (128x128 bf16)
// 1 block per t, 8 waves.  S tile staged to LDS by the Tensor Data Mover
// (fallback: async-load-to-LDS, then plain copy), transposed once in LDS so
// all fragments are contiguous ds_load_b128; A tile converted to bf16 in LDS.
// 64 KB LDS / block.
// ---------------------------------------------------------------------------
__global__ __launch_bounds__(256) void
slam_bmm_stage(const float* __restrict__ x_main_last,
               const __bf16* __restrict__ S_perm, float* __restrict__ out)
{
  __shared__ __bf16 u_lds[MM * MM];    // 32 KB: S raw (j,k), later A bf16 tile
  __shared__ __bf16 bt_lds[MM * MM];   // 32 KB: S transposed (k,j)
  const int t    = blockIdx.x;
  const int lane = threadIdx.x & 31;
  const int wave = threadIdx.x >> 5;
  const int hi   = lane >> 4;
  const int lrow = lane & 15;

  // ---- 1) S_perm[t] -> u_lds (raw, row-major (j,k)) ----
#if defined(HAVE_TDM)
  if (wave == 0) {
    unsigned long long ga =
        (unsigned long long)(const void*)(S_perm + (size_t)t * (MM * MM));
    unsigned int lds_addr = (unsigned int)(unsigned long long)(const void*)u_lds;
    // D# group 0: count=1 | lds_addr | global_addr[56:0] | type=2
    v4u_raw g0;
    g0[0] = 1u;                                  // count=1 (valid user D#)
    g0[1] = lds_addr;                            // lds_addr (bytes)
    g0[2] = (unsigned int)(ga & 0xffffffffull);  // global_addr[31:0]
    g0[3] = (unsigned int)((ga >> 32) & 0x01ffffffull) | 0x80000000u;  // type=2
    // D# group 1: data_size=2B; tensor 128x128; tile 128x128; dim0 stride 128
    v8i_raw g1;
    g1[0] = 1 << 16;       // workgroup_mask=0, data_size=1 (2 bytes)
    g1[1] = MM << 16;      // tensor_dim0[15:0] in bits[79:64->31:16]
    g1[2] = MM << 16;      // tensor_dim0 hi=0 | tensor_dim1[15:0]
    g1[3] = MM << 16;      // tensor_dim1 hi=0 | tile_dim0=128
    g1[4] = MM;            // tile_dim1=128, tile_dim2=0
    g1[5] = MM;            // tensor_dim0_stride[31:0] = 128
    g1[6] = 0;             // stride hi | tensor_dim1_stride lo
    g1[7] = 0;
    v4i_raw g2 = {0, 0, 0, 0};   // 2-D tensor: groups 2/3 unused
    v4i_raw g3 = {0, 0, 0, 0};
    v8i_raw g4 = {0, 0, 0, 0, 0, 0, 0, 0};  // 6-arg toolchain: extra group
    __builtin_amdgcn_tensor_load_to_lds(g0, g1, g2, g3, g4, 0);
#if __has_builtin(__builtin_amdgcn_s_wait_tensorcnt)
    __builtin_amdgcn_s_wait_tensorcnt(0);
#else
    asm volatile("s_wait_tensorcnt 0x0" ::: "memory");
#endif
  }
#elif defined(HAVE_ASYNC_LDS)
  {
    unsigned long long g =
        (unsigned long long)(const void*)(S_perm + (size_t)t * (MM * MM));
    unsigned int l = (unsigned int)(unsigned long long)(const void*)u_lds;
    for (int i = (int)threadIdx.x; i < 2048; i += 256) {
      __builtin_amdgcn_global_load_async_to_lds_b128(
          (gv4i_p)(g + (unsigned long long)i * 16),
          (lv4i_p)(l + (unsigned int)i * 16),
          0, 0);
    }
    asm volatile("s_wait_asynccnt 0x0" ::: "memory");
  }
#else
  {
    const uint4* src = (const uint4*)(S_perm + (size_t)t * (MM * MM));
    uint4* dst = (uint4*)u_lds;
#pragma unroll
    for (int i = 0; i < 8; ++i)
      dst[threadIdx.x + i * 256] = src[threadIdx.x + i * 256];
  }
#endif
  __syncthreads();

  // ---- 2) transpose u_lds (j,k) -> bt_lds (k,j) ----
#pragma unroll
  for (int c = 0; c < 8; ++c) {
    int base = c * 2048 + (int)threadIdx.x * 8;  // 8 consecutive k at fixed j
    int j = base >> 7, k0 = base & 127;
    union { __bf16 h[8]; uint4 q; } p;
    p.q = *(const uint4*)&u_lds[base];
#pragma unroll
    for (int e = 0; e < 8; ++e) bt_lds[(k0 + e) * MM + j] = p.h[e];
  }
  __syncthreads();

  // ---- 3) A tile f32 -> bf16 into u_lds (overwrite) ----
  const float* As = x_main_last + (size_t)t * (MM * MM);
  __builtin_prefetch(As + threadIdx.x * 4, 0, 1);   // global_prefetch_b8
#pragma unroll
  for (int c = 0; c < 16; ++c) {
    int idx = c * 1024 + (int)threadIdx.x * 4;
    float4 f = *(const float4*)(As + idx);
    union { __bf16 h[4]; uint2 u; } p;
    p.h[0] = (__bf16)f.x; p.h[1] = (__bf16)f.y;
    p.h[2] = (__bf16)f.z; p.h[3] = (__bf16)f.w;
    *(uint2*)&u_lds[idx] = p.u;
  }
  __syncthreads();

  // ---- 4) fragments + WMMA ----
  v16bf afr[4];
  const __bf16* arow = &u_lds[(wave * 16 + lrow) * MM];
#pragma unroll
  for (int kb = 0; kb < 4; ++kb) {
    const __bf16* ap = arow + kb * 32 + hi * 8;
    afr[kb] = frag_from(ap, ap + 16);
  }

  v8f acc[8];
#pragma unroll
  for (int n = 0; n < 8; ++n)
#pragma unroll
    for (int i = 0; i < 8; ++i) acc[n][i] = 0.f;

#pragma unroll
  for (int n = 0; n < 8; ++n) {
#pragma unroll
    for (int kb = 0; kb < 4; ++kb) {
      const __bf16* bp = &bt_lds[(n * 16 + lrow) * MM + kb * 32 + hi * 8];
      acc[n] = wmma_bf16(afr[kb], frag_from(bp, bp + 16), acc[n]);
    }
  }

  // ---- 5) store f32 output ----
  float* orow = out + (size_t)t * (MM * MM);
#pragma unroll
  for (int n = 0; n < 8; ++n) {
#pragma unroll
    for (int i = 0; i < 8; ++i) {
      int r = wave * 16 + i + (hi << 3);
      orow[r * MM + n * 16 + lrow] = acc[n][i];
    }
  }
}

extern "C" void kernel_launch(void* const* d_in, const int* in_sizes, int n_in,
                              void* d_out, int out_size, void* d_ws, size_t ws_size,
                              hipStream_t stream) {
  const float* x_main = (const float*)d_in[0];  // (4, 1024, 1, 128, 128)
  const float* x_aux  = (const float*)d_in[1];  // (128, 1024, 64)
  const float* W1     = (const float*)d_in[2];  // (64, 128)
  const float* b1     = (const float*)d_in[3];  // (128,)
  const float* W2     = (const float*)d_in[4];  // (128, 128)
  const float* b2     = (const float*)d_in[5];  // (128,)
  float* out = (float*)d_out;                   // (1, 1024, 128, 128, 1)

  __bf16* S_perm = (__bf16*)d_ws;                       // 33.5 MB
  __bf16* W1t    = S_perm + (size_t)TT * MM * MM;       // 16 KB
  __bf16* W2t    = W1t + (size_t)MM * AA;               // 32 KB

  const float* x_main_last = x_main + (size_t)3 * TT * MM * MM;  // x_main[-1,:,0]

  prep_weights<<<1, 128, 0, stream>>>(W1, W2, W1t, W2t);
  slam_mlp_stage<<<dim3(TT / 16, MM / 4), 128, 0, stream>>>(x_aux, b1, b2, W1t,
                                                            W2t, S_perm);
  slam_bmm_stage<<<TT, 256, 0, stream>>>(x_main_last, S_perm, out);
}
